// BiMambaBlock_3375844295413
// MI455X (gfx1250) — matrix operand order repaired and checked by reading, not compiled, not run on hardware
//
#include <hip/hip_runtime.h>
#include <hip/hip_bf16.h>
#include <math.h>

// ---------------------------------------------------------------- constants
#define BB 4
#define LL 2048
#define CC 512
#define DI 1024      // D_INNER
#define DS 16        // D_STATE
#define DR 32        // DT_RANK
#define ML (BB*LL)   // 8192 rows per direction

typedef __attribute__((ext_vector_type(16))) __bf16 bf16x16;
typedef __attribute__((ext_vector_type(8)))  __bf16 bf16x8;
typedef __attribute__((ext_vector_type(8)))  float  f32x8;
typedef unsigned int u32x4 __attribute__((ext_vector_type(4)));
typedef int          i32x4 __attribute__((ext_vector_type(4)));
typedef int          i32x8 __attribute__((ext_vector_type(8)));

// Use the Tensor Data Mover only in the device pass and only if the builtin
// exists in this toolchain; otherwise fall back to load+ds_store staging.
#if defined(__HIP_DEVICE_COMPILE__) && __has_builtin(__builtin_amdgcn_tensor_load_to_lds)
#define HAVE_TDM 1
#else
#define HAVE_TDM 0
#endif

// ---------------------------------------------------------------- helpers
__device__ __forceinline__ float act_apply(float v, int act) {
  if (act == 1) {                       // softplus
    return (v > 20.f) ? v : log1pf(__expf(v));
  } else if (act == 2) {                // exact GELU
    return 0.5f * v * (1.f + erff(v * 0.70710678118654752f));
  }
  return v;
}

__device__ __forceinline__ float silu(float v) {
  return v / (1.f + __expf(-v));
}

#if HAVE_TDM
// Issue one TDM tile load: 64 rows x 32 bf16 cols from a [.., lda] bf16 matrix
// into LDS with 16B padding per 64B row (row stride 80B = 40 elements).
// D# layout per CDNA5 ISA 8.3/8.4 (group0: flags/lds/global addr; group1: dims).
__device__ __forceinline__ void tdm_load_tile(unsigned lds_addr,
                                              const __bf16* gptr, int lda_elems) {
  unsigned long long ga = (unsigned long long)(size_t)gptr;
  u32x4 g0 = {};
  g0[0] = 1u;                                   // count=1 (valid), user mode
  g0[1] = lds_addr;                             // LDS byte address
  g0[2] = (unsigned)ga;                         // global_addr[31:0]
  g0[3] = (unsigned)(ga >> 32) | 0x80000000u;   // global_addr[56:32] | type=2
  i32x8 g1 = {};
  // data_size=1(2B) | pad_enable | pad_interval=3(16 DW=64B) | pad_amount=3(4 DW=16B)
  g1[0] = (int)0x06D10000u;
  g1[1] = (int)(32u << 16);                     // tensor_dim0 = 32 (lo16 in [63:48])
  g1[2] = (int)(64u << 16);                     // tensor_dim1 = 64 (lo16 in [95:80])
  g1[3] = (int)(32u << 16);                     // tile_dim0 = 32   ([127:112])
  g1[4] = 64;                                   // tile_dim1 = 64   ([143:128])
  g1[5] = lda_elems;                            // tensor_dim0_stride[31:0]
  g1[6] = 0;
  g1[7] = 0;
  i32x4 z4 = {};
#if defined(__clang_major__) && (__clang_major__ >= 23)
  i32x8 z8 = {};
  __builtin_amdgcn_tensor_load_to_lds(g0, g1, z4, z4, z8, 0);
#else
  __builtin_amdgcn_tensor_load_to_lds(g0, g1, z4, z4, 0);
#endif
}
#endif

// ---------------------------------------------------------------- f32 -> bf16
__global__ void k_f32_to_bf16(const float* __restrict__ in, __bf16* __restrict__ out, long n) {
  long i = (long)blockIdx.x * blockDim.x + threadIdx.x;
  long stride = (long)gridDim.x * blockDim.x;
  for (; i < n; i += stride) out[i] = (__bf16)in[i];
}

// ---------------------------------------------------------------- LayerNorm
__global__ __launch_bounds__(256)
void k_ln_dual(const float* __restrict__ x, const float* __restrict__ g,
               const float* __restrict__ bta, __bf16* __restrict__ out0,
               __bf16* __restrict__ out1) {
  __shared__ float red[256];
  const int row = blockIdx.x;                 // 0..ML-1
  const int tid = threadIdx.x;
  const float* xr = x + (long)row * CC;

  float s = 0.f;
  for (int i = tid; i < CC; i += 256) s += xr[i];
  red[tid] = s; __syncthreads();
  for (int o = 128; o > 0; o >>= 1) { if (tid < o) red[tid] += red[tid + o]; __syncthreads(); }
  const float mu = red[0] * (1.f / CC);
  __syncthreads();

  float v = 0.f;
  for (int i = tid; i < CC; i += 256) { float d = xr[i] - mu; v += d * d; }
  red[tid] = v; __syncthreads();
  for (int o = 128; o > 0; o >>= 1) { if (tid < o) red[tid] += red[tid + o]; __syncthreads(); }
  const float rstd = rsqrtf(red[0] * (1.f / CC) + 1e-5f);

  const int b = row / LL, l = row % LL;
  const long row1 = (long)b * LL + (LL - 1 - l);
  for (int i = tid; i < CC; i += 256) {
    float y = (xr[i] - mu) * rstd * g[i] + bta[i];
    out0[(long)row * CC + i] = (__bf16)y;
    if (out1) out1[row1 * CC + i] = (__bf16)y;
  }
}

// ---------------------------------------------------------------- WMMA GEMM
// Cout[z][m][n] = act( sum_k A[z][m][k]*W[z][n][k] + bias[z][n] )
// 256 threads = 8 waves, block tile 64x64, K step 32, double-buffered LDS.
// TDM path: wave 0 programs D# and the Tensor Data Mover DMAs each 64x32 tile
// (with HW padding to an 80B row stride) while the other buffer is computed.
__global__ __launch_bounds__(256)
void k_gemm_bf16(const __bf16* __restrict__ A, long strideA,
                 const __bf16* __restrict__ W, long strideW,
                 float* __restrict__ Cout, long strideC,
                 const float* __restrict__ bias, long strideBias,
                 int M, int N, int K, int lda, int act) {
  const int bm = blockIdx.x * 64;
  const int bn = blockIdx.y * 64;
  const int z  = blockIdx.z;
  A    += (long)z * strideA;
  W    += (long)z * strideW;
  Cout += (long)z * strideC;
  const float* bp = bias ? bias + (long)z * strideBias : (const float*)0;

  __shared__ __attribute__((aligned(16))) __bf16 As[2][64][40];  // 80B row stride
  __shared__ __attribute__((aligned(16))) __bf16 Bs[2][64][40];

  const int tid  = threadIdx.x;
  const int lane = tid & 31;
  const int w    = tid >> 5;
  const int half = lane >> 4;
  const int l15  = lane & 15;
  const int mo   = 16 * (w >> 1);   // wave M offset: 0/16/32/48
  const int no   = 32 * (w & 1);    // wave N offset: 0/32

  f32x8 c0 = {}; f32x8 c1 = {};

#if HAVE_TDM
  const __bf16* Ablk = A + (long)bm * lda;
  const __bf16* Wblk = W + (long)bn * K;
  const unsigned ldsA0 = (unsigned)(size_t)&As[0][0][0];
  const unsigned ldsA1 = (unsigned)(size_t)&As[1][0][0];
  const unsigned ldsB0 = (unsigned)(size_t)&Bs[0][0][0];
  const unsigned ldsB1 = (unsigned)(size_t)&Bs[1][0][0];
  const bool issuer = (w == 0);

  if (issuer) {                       // prime buffer 0
    tdm_load_tile(ldsA0, Ablk, lda);
    tdm_load_tile(ldsB0, Wblk, K);
  }
  int buf = 0;
  for (int kk = 0; kk < K; kk += 32, buf ^= 1) {
    if (issuer) {
      if (kk + 32 < K) {              // overlap next-stage DMA with compute
        tdm_load_tile(buf ? ldsA0 : ldsA1, Ablk + kk + 32, lda);
        tdm_load_tile(buf ? ldsB0 : ldsB1, Wblk + kk + 32, K);
        __builtin_amdgcn_s_wait_tensorcnt(2);   // in-order: current stage done
      } else {
        __builtin_amdgcn_s_wait_tensorcnt(0);
      }
    }
    __syncthreads();

    bf16x8 a0 = *(const bf16x8*)&As[buf][mo + l15][16 * half];
    bf16x8 a1 = *(const bf16x8*)&As[buf][mo + l15][16 * half + 8];
    bf16x16 af = __builtin_shufflevector(a0, a1, 0,1,2,3,4,5,6,7,8,9,10,11,12,13,14,15);
    bf16x8 b00 = *(const bf16x8*)&Bs[buf][no + l15][16 * half];
    bf16x8 b01 = *(const bf16x8*)&Bs[buf][no + l15][16 * half + 8];
    bf16x16 bf0 = __builtin_shufflevector(b00, b01, 0,1,2,3,4,5,6,7,8,9,10,11,12,13,14,15);
    bf16x8 b10 = *(const bf16x8*)&Bs[buf][no + 16 + l15][16 * half];
    bf16x8 b11 = *(const bf16x8*)&Bs[buf][no + 16 + l15][16 * half + 8];
    bf16x16 bf1 = __builtin_shufflevector(b10, b11, 0,1,2,3,4,5,6,7,8,9,10,11,12,13,14,15);

    c0 = __builtin_amdgcn_wmma_f32_16x16x32_bf16(false, af, false, bf0, (short)0, c0, false, false);
    c1 = __builtin_amdgcn_wmma_f32_16x16x32_bf16(false, af, false, bf1, (short)0, c1, false, false);
    __syncthreads();
  }
#else
  // Fallback: cooperative load + ds_store staging (proven path)
  const int lrow   = tid >> 2;        // 0..63
  const int lchunk = tid & 3;         // 0..3
  for (int kk = 0; kk < K; kk += 32) {
    uint4 va = *(const uint4*)(A + (long)(bm + lrow) * lda + kk + 8 * lchunk);
    uint4 vb = *(const uint4*)(W + (long)(bn + lrow) * K   + kk + 8 * lchunk);
    if (kk + 32 < K) {
      __builtin_prefetch((const void*)(A + (long)(bm + lrow) * lda + kk + 32 + 8 * lchunk), 0, 1);
      __builtin_prefetch((const void*)(W + (long)(bn + lrow) * K   + kk + 32 + 8 * lchunk), 0, 1);
    }
    *(uint4*)&As[0][lrow][8 * lchunk] = va;
    *(uint4*)&Bs[0][lrow][8 * lchunk] = vb;
    __syncthreads();

    bf16x8 a0 = *(const bf16x8*)&As[0][mo + l15][16 * half];
    bf16x8 a1 = *(const bf16x8*)&As[0][mo + l15][16 * half + 8];
    bf16x16 af = __builtin_shufflevector(a0, a1, 0,1,2,3,4,5,6,7,8,9,10,11,12,13,14,15);
    bf16x8 b00 = *(const bf16x8*)&Bs[0][no + l15][16 * half];
    bf16x8 b01 = *(const bf16x8*)&Bs[0][no + l15][16 * half + 8];
    bf16x16 bf0 = __builtin_shufflevector(b00, b01, 0,1,2,3,4,5,6,7,8,9,10,11,12,13,14,15);
    bf16x8 b10 = *(const bf16x8*)&Bs[0][no + 16 + l15][16 * half];
    bf16x8 b11 = *(const bf16x8*)&Bs[0][no + 16 + l15][16 * half + 8];
    bf16x16 bf1 = __builtin_shufflevector(b10, b11, 0,1,2,3,4,5,6,7,8,9,10,11,12,13,14,15);

    c0 = __builtin_amdgcn_wmma_f32_16x16x32_bf16(false, af, false, bf0, (short)0, c0, false, false);
    c1 = __builtin_amdgcn_wmma_f32_16x16x32_bf16(false, af, false, bf1, (short)0, c1, false, false);
    __syncthreads();
  }
#endif

  // Epilogue: D layout -> VGPR r holds row (r + 8*half), col = lane&15
  const int col0 = bn + no + l15;
  const int col1 = bn + no + 16 + l15;
  float bias0 = bp ? bp[col0] : 0.f;
  float bias1 = bp ? bp[col1] : 0.f;
#pragma unroll
  for (int r = 0; r < 8; ++r) {
    int row = bm + mo + r + 8 * half;
    Cout[(long)row * N + col0] = act_apply(c0[r] + bias0, act);
    Cout[(long)row * N + col1] = act_apply(c1[r] + bias1, act);
  }
}

// ---------------------------------------------------------------- causal depthwise conv + SiLU
__global__ void k_conv_silu(const float* __restrict__ uz, const float* __restrict__ cw,
                            const float* __restrict__ cb, float* __restrict__ uc,
                            __bf16* __restrict__ uc_bf) {
  long idx = (long)blockIdx.x * blockDim.x + threadIdx.x;
  const long total = 2L * ML * DI;
  long stride = (long)gridDim.x * blockDim.x;
  for (; idx < total; idx += stride) {
    int  ch = (int)(idx % DI);
    long t  = idx / DI;
    int  l  = (int)(t % LL);
    long db = t / LL;            // dir*B + b
    int  d  = (int)(db >> 2);    // B == 4
    const float* wp = cw + ((long)d * DI + ch) * 4;
    float acc = cb[(long)d * DI + ch];
#pragma unroll
    for (int j = 0; j < 4; ++j) {
      int lp = l - 3 + j;
      if (lp >= 0) acc += uz[((long)db * LL + lp) * (2 * DI) + ch] * wp[j];
    }
    float sv = silu(acc);
    uc[idx] = sv;
    uc_bf[idx] = (__bf16)sv;
  }
}

// ---------------------------------------------------------------- selective scan
__global__ __launch_bounds__(256)
void k_scan(const float* __restrict__ dt, const float* __restrict__ xdbl,
            const float* __restrict__ ucv, const float* __restrict__ A_log,
            float* __restrict__ ys) {
  const int db = blockIdx.x;                 // dir*B + b
  const int d  = db >> 2;                    // B == 4
  const int ch = blockIdx.y * 256 + threadIdx.x;

  float a[DS], h[DS];
#pragma unroll
  for (int s = 0; s < DS; ++s) {
    a[s] = -__expf(A_log[((long)d * DI + ch) * DS + s]);
    h[s] = 0.f;
  }

  __shared__ float sBC[2 * DS];
  for (int l = 0; l < LL; ++l) {
    const long row = (long)db * LL + l;
    if (threadIdx.x < 2 * DS) sBC[threadIdx.x] = xdbl[row * 64 + DR + threadIdx.x];
    __syncthreads();
    float dtv = dt[row * DI + ch];
    float du  = dtv * ucv[row * DI + ch];
    float y = 0.f;
#pragma unroll
    for (int s = 0; s < DS; ++s) {
      h[s] = __expf(dtv * a[s]) * h[s] + du * sBC[s];
      y += h[s] * sBC[DS + s];
    }
    ys[row * DI + ch] = y;
    __syncthreads();
  }
}

// ---------------------------------------------------------------- gate
__global__ void k_gate(const float* __restrict__ uz, const float* __restrict__ ys,
                       const float* __restrict__ ucv, const float* __restrict__ Dp,
                       __bf16* __restrict__ gated) {
  long idx = (long)blockIdx.x * blockDim.x + threadIdx.x;
  const long total = 2L * ML * DI;
  long stride = (long)gridDim.x * blockDim.x;
  for (; idx < total; idx += stride) {
    int  ch = (int)(idx % DI);
    long t  = idx / DI;
    int  l  = (int)(t % LL);
    long db = t / LL;
    int  d  = (int)(db >> 2);
    float zv = uz[((long)db * LL + l) * (2 * DI) + DI + ch];
    float y  = ys[idx] + ucv[idx] * Dp[(long)d * DI + ch];
    gated[idx] = (__bf16)(y * silu(zv));
  }
}

// ---------------------------------------------------------------- combine + residual
__global__ void k_combine(const float* __restrict__ x, const float* __restrict__ m0,
                          const float* __restrict__ m1, float* __restrict__ x1) {
  long idx = (long)blockIdx.x * blockDim.x + threadIdx.x;
  const long total = (long)ML * CC;
  long stride = (long)gridDim.x * blockDim.x;
  for (; idx < total; idx += stride) {
    int  c = (int)(idx % CC);
    long t = idx / CC;
    int  l = (int)(t % LL);
    long b = t / LL;
    float v1 = m1[((long)b * LL + (LL - 1 - l)) * CC + c];
    x1[idx] = x[idx] + 0.5f * (m0[idx] + v1);
  }
}

// ---------------------------------------------------------------- final residual
__global__ void k_add(const float* __restrict__ a, const float* __restrict__ b,
                      float* __restrict__ o, long n) {
  long idx = (long)blockIdx.x * blockDim.x + threadIdx.x;
  long stride = (long)gridDim.x * blockDim.x;
  for (; idx < n; idx += stride) o[idx] = a[idx] + b[idx];
}

// ---------------------------------------------------------------- launch
extern "C" void kernel_launch(void* const* d_in, const int* in_sizes, int n_in,
                              void* d_out, int out_size, void* d_ws, size_t ws_size,
                              hipStream_t stream) {
  (void)in_sizes; (void)n_in; (void)out_size; (void)ws_size;

  const float* x      = (const float*)d_in[0];
  const float* gamma1 = (const float*)d_in[1];
  const float* beta1  = (const float*)d_in[2];
  const float* W_in   = (const float*)d_in[3];
  const float* conv_w = (const float*)d_in[4];
  const float* conv_b = (const float*)d_in[5];
  const float* W_xp   = (const float*)d_in[6];
  const float* W_dt   = (const float*)d_in[7];
  const float* b_dt   = (const float*)d_in[8];
  const float* A_log  = (const float*)d_in[9];
  const float* Dp     = (const float*)d_in[10];
  const float* W_out  = (const float*)d_in[11];
  const float* gamma2 = (const float*)d_in[12];
  const float* beta2  = (const float*)d_in[13];
  const float* W1     = (const float*)d_in[14];
  const float* b1     = (const float*)d_in[15];
  const float* W2     = (const float*)d_in[16];
  const float* b2     = (const float*)d_in[17];
  float* out = (float*)d_out;

  size_t off = 0;
  auto alloc = [&](size_t bytes) -> void* {
    off = (off + 255) & ~(size_t)255;
    void* p = (char*)d_ws + off;
    off += bytes;
    return p;
  };

  __bf16* xn2_bf  = (__bf16*)alloc(2L * ML * CC * 2);
  __bf16* Wb_in   = (__bf16*)alloc(2L * 2 * DI * CC * 2);
  __bf16* Wb_xp   = (__bf16*)alloc(2L * 64 * DI * 2);
  __bf16* Wb_dt   = (__bf16*)alloc(2L * DI * DR * 2);
  __bf16* Wb_out  = (__bf16*)alloc(2L * CC * DI * 2);
  __bf16* Wb_1    = (__bf16*)alloc((long)(4 * CC) * CC * 2);
  __bf16* Wb_2    = (__bf16*)alloc((long)CC * (4 * CC) * 2);
  float*  uz      = (float*)alloc(2L * ML * (2 * DI) * 4);
  float*  ucv     = (float*)alloc(2L * ML * DI * 4);
  __bf16* ucv_bf  = (__bf16*)alloc(2L * ML * DI * 2);
  float*  xdbl    = (float*)alloc(2L * ML * 64 * 4);
  __bf16* xdbl_bf = (__bf16*)alloc(2L * ML * 64 * 2);
  float*  dtb     = (float*)alloc(2L * ML * DI * 4);
  float*  ysb     = (float*)alloc(2L * ML * DI * 4);
  __bf16* gated   = (__bf16*)alloc(2L * ML * DI * 2);
  float*  mout    = (float*)alloc(2L * ML * CC * 4);
  float*  x1      = (float*)alloc((long)ML * CC * 4);
  // MLP phase reuses dead buffers:
  __bf16* ln2_bf = xn2_bf;
  float*  hbuf   = uz;
  __bf16* h_bf   = ucv_bf;
  float*  mlp    = dtb;

  auto cvt = [&](const float* src, __bf16* dst, long n) {
    int blocks = (int)((n + 2047) / 2048);
    k_f32_to_bf16<<<blocks, 256, 0, stream>>>(src, dst, n);
  };
  auto gemm = [&](const __bf16* A, long sA, int lda, const __bf16* W, long sW,
                  float* Cp, long sC, const float* bias, long sBias,
                  int M, int N, int K, int act, int batch) {
    dim3 g(M / 64, N / 64, batch);
    k_gemm_bf16<<<g, 256, 0, stream>>>(A, sA, W, sW, Cp, sC, bias, sBias, M, N, K, lda, act);
  };

  // 0) weights -> bf16
  cvt(W_in,  Wb_in,  2L * 2 * DI * CC);
  cvt(W_xp,  Wb_xp,  2L * 64 * DI);
  cvt(W_dt,  Wb_dt,  2L * DI * DR);
  cvt(W_out, Wb_out, 2L * CC * DI);
  cvt(W1,    Wb_1,   (long)(4 * CC) * CC);
  cvt(W2,    Wb_2,   (long)CC * (4 * CC));

  // 1) LN1 -> bf16 rows for both directions (dir1 L-flipped)
  k_ln_dual<<<ML, 256, 0, stream>>>(x, gamma1, beta1, xn2_bf, xn2_bf + (long)ML * CC);

  // 2) in-proj: [8192x512] @ [2048x512]^T -> uz, batched over 2 dirs
  gemm(xn2_bf, (long)ML * CC, CC, Wb_in, 2L * DI * CC,
       uz, (long)ML * 2 * DI, nullptr, 0, ML, 2 * DI, CC, 0, 2);

  // 3) causal depthwise conv + SiLU
  {
    long n = 2L * ML * DI;
    k_conv_silu<<<(int)((n + 255) / 256), 256, 0, stream>>>(uz, conv_w, conv_b, ucv, ucv_bf);
  }

  // 4) x-proj: [8192x1024] @ [64x1024]^T -> x_dbl
  gemm(ucv_bf, (long)ML * DI, DI, Wb_xp, 64L * DI,
       xdbl, (long)ML * 64, nullptr, 0, ML, 64, DI, 0, 2);
  cvt(xdbl, xdbl_bf, 2L * ML * 64);

  // 5) dt: [8192x32(lda=64)] @ [1024x32]^T + b_dt, softplus
  gemm(xdbl_bf, (long)ML * 64, 64, Wb_dt, (long)DI * DR,
       dtb, (long)ML * DI, b_dt, DI, ML, DI, DR, 1, 2);

  // 6) selective scan
  k_scan<<<dim3(2 * BB, DI / 256), 256, 0, stream>>>(dtb, xdbl, ucv, A_log, ysb);

  // 7) gate: (y + u*Dp) * silu(z) -> bf16
  {
    long n = 2L * ML * DI;
    k_gate<<<(int)((n + 255) / 256), 256, 0, stream>>>(uz, ysb, ucv, Dp, gated);
  }

  // 8) out-proj: [8192x1024] @ [512x1024]^T
  gemm(gated, (long)ML * DI, DI, Wb_out, (long)CC * DI,
       mout, (long)ML * CC, nullptr, 0, ML, CC, DI, 0, 2);

  // 9) combine directions + residual
  {
    long n = (long)ML * CC;
    k_combine<<<(int)((n + 255) / 256), 256, 0, stream>>>(x, mout, mout + (long)ML * CC, x1);
  }

  // 10) LN2 -> bf16
  k_ln_dual<<<ML, 256, 0, stream>>>(x1, gamma2, beta2, ln2_bf, nullptr);

  // 11) MLP up: [8192x512] @ [2048x512]^T + b1, GELU
  gemm(ln2_bf, 0, CC, Wb_1, 0, hbuf, 0, b1, 0, ML, 4 * CC, CC, 2, 1);
  cvt(hbuf, h_bf, (long)ML * 4 * CC);

  // 12) MLP down: [8192x2048] @ [512x2048]^T + b2
  gemm(h_bf, 0, 4 * CC, Wb_2, 0, mlp, 0, b2, 0, ML, CC, 4 * CC, 0, 1);

  // 13) final residual
  {
    long n = (long)ML * CC;
    k_add<<<(int)((n + 255) / 256), 256, 0, stream>>>(x1, mlp, out, n);
  }
}
